// FIRE_RED_DECODER_52793738003210
// MI455X (gfx1250) — compile-verified, matrix operand order checked by reading, not run on hardware
//
#include <hip/hip_runtime.h>
#include <hip/hip_bf16.h>

// FireRed-style decoder, single-token decode step (S=1), KV-cached.
// Bandwidth-bound (~3.4 GB @ 23.3 TB/s): cache copy fused into self-attention,
// weight GEMMs via v_wmma_f32_16x16x32_f16 (wave32), A zero-padded to M=16
// so the WMMA loop has no divergence, A loaded as b128, 2 N-tiles per wave.

typedef _Float16 v16h __attribute__((ext_vector_type(16)));
typedef float    v8f  __attribute__((ext_vector_type(8)));

#define LL    16
#define BB    4
#define HH    20
#define DHH   64
#define DD    1280
#define DFFF  5120
#define VV    8000
#define HISTC 1024
#define TENCC 512
#define KVC   1025   // HIST + S

// ---------------------------------------------------------------------------
// GEMM: one wave computes a 16x32 output tile (two WMMA 16x16 tiles).
// A is (16, K) zero-padded fp32 in workspace (rows >= 4 are zero).
// out[m, n] = op( sum_k A[m,k] * W[y*wy + k*ldw + n] + bias[y*biasy + n] )
// mode: 0 = store, 1 = add-into (residual), 2 = relu-store
// ---------------------------------------------------------------------------
__global__ __launch_bounds__(32)
void gemm16_wmma_kernel(const float* __restrict__ A, int lda, int K,
                        const float* __restrict__ W, long long wy, int ldw,
                        const float* __restrict__ bias, long long biasy,
                        float* __restrict__ out, long long ob, long long oy,
                        long long oc, int mode)
{
    const int lane  = threadIdx.x;        // 0..31, wave32
    const int l15   = lane & 15;          // M row (A) / N col (B)
    const int khalf = (lane >> 4) * 8;    // lanes 16-31: K+8 / K+24 halves
    const int nb = blockIdx.x * 32;
    const int y  = blockIdx.y;
    const float* Wp = W + (long long)y * wy + nb;
    const float* Ap = A + l15 * lda + khalf;   // contiguous in k per lane

    v8f c0 = {0.f, 0.f, 0.f, 0.f, 0.f, 0.f, 0.f, 0.f};
    v8f c1 = {0.f, 0.f, 0.f, 0.f, 0.f, 0.f, 0.f, 0.f};

    for (int kb = 0; kb < K; kb += 32) {
        if (kb + 32 < K)  // prefetch next weight rows (global_prefetch_b8)
            __builtin_prefetch(Wp + (long long)(kb + 32) * ldw + l15, 0, 1);

        // ---- A fragment: two 8-float contiguous runs -> b128 loads ----
        const float4* ap = (const float4*)(Ap + kb);   // 16B aligned
        const float4 a0 = ap[0], a1 = ap[1];           // K = kb+khalf+0..7
        const float4 a2 = ap[4], a3 = ap[5];           // K = kb+16+khalf+0..7
        v16h af;
        af[0]  = (_Float16)a0.x; af[1]  = (_Float16)a0.y;
        af[2]  = (_Float16)a0.z; af[3]  = (_Float16)a0.w;
        af[4]  = (_Float16)a1.x; af[5]  = (_Float16)a1.y;
        af[6]  = (_Float16)a1.z; af[7]  = (_Float16)a1.w;
        af[8]  = (_Float16)a2.x; af[9]  = (_Float16)a2.y;
        af[10] = (_Float16)a2.z; af[11] = (_Float16)a2.w;
        af[12] = (_Float16)a3.x; af[13] = (_Float16)a3.y;
        af[14] = (_Float16)a3.z; af[15] = (_Float16)a3.w;

        // ---- B fragments: lane gathers column n across 16 k rows ----
        v16h bf0, bf1;
        #pragma unroll
        for (int e = 0; e < 8; ++e) {
            const long long k0 = (long long)(kb + khalf + e) * ldw;
            const long long k1 = (long long)(kb + 16 + khalf + e) * ldw;
            bf0[e]     = (_Float16)Wp[k0 + l15];
            bf0[e + 8] = (_Float16)Wp[k1 + l15];
            bf1[e]     = (_Float16)Wp[k0 + 16 + l15];
            bf1[e + 8] = (_Float16)Wp[k1 + 16 + l15];
        }
        c0 = __builtin_amdgcn_wmma_f32_16x16x32_f16(
                 false, af, false, bf0, (short)0, c0, false, false);
        c1 = __builtin_amdgcn_wmma_f32_16x16x32_f16(
                 false, af, false, bf1, (short)0, c1, false, false);
    }

    if (lane < 16) {
        #pragma unroll
        for (int t2 = 0; t2 < 2; ++t2) {
            const int n = nb + t2 * 16 + l15;
            const float bv = bias ? bias[(long long)y * biasy + n] : 0.f;
            #pragma unroll
            for (int m = 0; m < BB; ++m) {
                float val = (t2 ? c1[m] : c0[m]) + bv;  // lanes 0-15: vgpr r -> M=r
                float* o = out + (long long)m * ob + (long long)y * oy
                               + (long long)n * oc;
                if (mode == 1)      *o += val;
                else if (mode == 2) *o = val > 0.f ? val : 0.f;
                else                *o = val;
            }
        }
    }
}

// ---------------------------------------------------------------------------
// LayerNorm over D per batch row (writes rows 0..B-1 of a 16-row buffer).
// ---------------------------------------------------------------------------
__global__ __launch_bounds__(256)
void ln_kernel(const float* __restrict__ x, const float* __restrict__ w,
               const float* __restrict__ bb, float* __restrict__ out, int D)
{
    __shared__ float red[256];
    const int b = blockIdx.x, t = threadIdx.x;
    const float* xb = x + (long long)b * D;

    float s = 0.f;
    for (int d = t; d < D; d += 256) s += xb[d];
    red[t] = s; __syncthreads();
    for (int o = 128; o > 0; o >>= 1) { if (t < o) red[t] += red[t + o]; __syncthreads(); }
    const float mean = red[0] / D; __syncthreads();

    float v = 0.f;
    for (int d = t; d < D; d += 256) { float df = xb[d] - mean; v += df * df; }
    red[t] = v; __syncthreads();
    for (int o = 128; o > 0; o >>= 1) { if (t < o) red[t] += red[t + o]; __syncthreads(); }
    const float rstd = rsqrtf(red[0] / D + 1e-5f);

    for (int d = t; d < D; d += 256)
        out[(long long)b * D + d] = (xb[d] - mean) * rstd * w[d] + bb[d];
}

__global__ void zero_kernel(float* __restrict__ p, long long n)
{
    const long long i = (long long)blockIdx.x * blockDim.x + threadIdx.x;
    if (i < n) p[i] = 0.f;
}

// ---------------------------------------------------------------------------
// Embedding + positional encoding.
// ---------------------------------------------------------------------------
__global__ void embed_kernel(const int* __restrict__ ids,
                             const float* __restrict__ emb,
                             const float* __restrict__ pe,
                             float* __restrict__ x, int hist)
{
    const int b = blockIdx.x;
    const int id = ids[b];                      // (B, S=1)
    const float* e = emb + (long long)id * DD;
    const float* p = pe + (long long)hist * DD; // pe[0, hist, :]
    for (int d = threadIdx.x; d < DD; d += blockDim.x)
        x[(long long)b * DD + d] = e[d] + p[d];
}

// ---------------------------------------------------------------------------
// Attention per (b,h). Self-attn (COPY=1) fuses the KV-cache copy into the
// score / weighted-sum streams so each cache element is read exactly once.
// JC = cached length, JT = total (JT = JC+1 for self; new column/row at JC
// was already written by the K/V GEMMs).
// ---------------------------------------------------------------------------
template<int JC, int JT, bool COPY, bool MASK>
__global__ __launch_bounds__(256)
void attn_kernel(const float* __restrict__ kc, const float* __restrict__ vc,
                 const float* __restrict__ q,
                 float* __restrict__ nk, float* __restrict__ nv,
                 float* __restrict__ ao, const float* __restrict__ mflagp)
{
    __shared__ float sq[DHH];
    __shared__ float sc[JT];
    __shared__ float red[256];
    __shared__ float acc4[256];

    const int h = blockIdx.x, b = blockIdx.y, t = threadIdx.x;
    const long long bh = (long long)b * HH + h;
    const float* kcb = kc + bh * DHH * JC;      // (DH, JC), j contiguous
    const float* vcb = vc + bh * JC * DHH;      // (JC, DH), d contiguous
    float* nkb = COPY ? nk + bh * DHH * JT : nullptr;
    float* nvb = COPY ? nv + bh * JT * DHH : nullptr;

    if (t < DHH) sq[t] = q[bh * DHH + t];
    __syncthreads();
    const float mflag = MASK ? mflagp[0] : 0.f;

    // scores (+ K copy)
    for (int j = t; j < JT; j += 256) {
        float s = 0.f;
        if (j < JC) {
            #pragma unroll 8
            for (int d = 0; d < DHH; ++d) {
                const float kv = kcb[(long long)d * JC + j];
                if (COPY) nkb[(long long)d * JT + j] = kv;
                s += sq[d] * kv;
            }
        } else {  // new token column, already in new_k
            #pragma unroll 8
            for (int d = 0; d < DHH; ++d) s += sq[d] * nkb[(long long)d * JT + j];
        }
        if (MASK && j >= 1) s += -128.f * mflag;  // tril mask, row 0 of S=1
        sc[j] = s;
    }
    __syncthreads();

    // softmax
    float mx = -1e30f;
    for (int j = t; j < JT; j += 256) mx = fmaxf(mx, sc[j]);
    red[t] = mx; __syncthreads();
    for (int o = 128; o > 0; o >>= 1) { if (t < o) red[t] = fmaxf(red[t], red[t + o]); __syncthreads(); }
    mx = red[0]; __syncthreads();
    float sm = 0.f;
    for (int j = t; j < JT; j += 256) { const float e = __expf(sc[j] - mx); sc[j] = e; sm += e; }
    red[t] = sm; __syncthreads();
    for (int o = 128; o > 0; o >>= 1) { if (t < o) red[t] += red[t + o]; __syncthreads(); }
    const float inv = 1.f / red[0]; __syncthreads();

    // weighted sum over V (+ V copy); 256 threads = 4 j-lanes x 64 d
    const int d = t & 63, jo = t >> 6;
    float a = 0.f;
    for (int j = jo; j < JT; j += 4) {
        float v;
        if (j < JC) {
            v = vcb[(long long)j * DHH + d];
            if (COPY) nvb[(long long)j * DHH + d] = v;
        } else {
            v = nvb[(long long)j * DHH + d];
        }
        a += sc[j] * v;
    }
    acc4[t] = a * inv; __syncthreads();
    if (jo == 0)
        ao[bh * DHH + d] = acc4[d] + acc4[64 + d] + acc4[128 + d] + acc4[192 + d];
}

__global__ void kv_scalar_kernel(const int* hl, const int* il, float* o)
{
    o[0] = (float)(hl[0] + il[0]);
}

// ---------------------------------------------------------------------------
extern "C" void kernel_launch(void* const* d_in, const int* in_sizes, int n_in,
                              void* d_out, int out_size, void* d_ws, size_t ws_size,
                              hipStream_t stream)
{
    const float* kcache  = (const float*)d_in[0];
    const float* vcache  = (const float*)d_in[1];
    const float* cross_k = (const float*)d_in[2];
    const float* cross_v = (const float*)d_in[3];
    const int*   ids     = (const int*)  d_in[4];
    const int*   hlen    = (const int*)  d_in[5];
    const int*   ilen    = (const int*)  d_in[6];
    const float* mflag   = (const float*)d_in[7];
    const float* emb     = (const float*)d_in[8];
    const float* pe      = (const float*)d_in[9];
    const float* snw     = (const float*)d_in[10];
    const float* snb     = (const float*)d_in[11];
    const float* wq      = (const float*)d_in[12];
    const float* bq      = (const float*)d_in[13];
    const float* wk      = (const float*)d_in[14];
    const float* wv      = (const float*)d_in[15];
    const float* bv      = (const float*)d_in[16];
    const float* fcw     = (const float*)d_in[17];
    const float* fcb     = (const float*)d_in[18];
    const float* cnw     = (const float*)d_in[19];
    const float* cnb     = (const float*)d_in[20];
    const float* cwq     = (const float*)d_in[21];
    const float* cbq     = (const float*)d_in[22];
    const float* cfcw    = (const float*)d_in[23];
    const float* cfcb    = (const float*)d_in[24];
    const float* mnw     = (const float*)d_in[25];
    const float* mnb     = (const float*)d_in[26];
    const float* w1      = (const float*)d_in[27];
    const float* b1      = (const float*)d_in[28];
    const float* w2      = (const float*)d_in[29];
    const float* b2      = (const float*)d_in[30];
    const float* lnow    = (const float*)d_in[31];
    const float* lnob    = (const float*)d_in[32];
    const float* proj_w  = (const float*)d_in[33];
    const float* proj_b  = (const float*)d_in[34];
    (void)in_sizes; (void)n_in; (void)out_size; (void)ws_size;

    float* out = (float*)d_out;
    const long long NK = (long long)LL * BB * HH * DHH * KVC;   // 83,968,000
    float* logits = out + 2 * NK;
    float* kvout  = logits + (long long)BB * VV;

    // ws layout: x (4,D) | qb (4,H,DH) | xn (16,D) | ab (16,H*DH) | tb (16,DFF)
    float* ws = (float*)d_ws;
    float* x  = ws;                       // 4 x 1280
    float* qb = x  + BB * DD;             // 4 x 1280
    float* xn = qb + BB * HH * DHH;       // 16 x 1280  (A operand, zero-padded)
    float* ab = xn + 16 * DD;             // 16 x 1280  (A operand, zero-padded)
    float* tb = ab + 16 * HH * DHH;       // 16 x 5120  (A operand, zero-padded)
    const long long padN = (long long)16 * DD + 16 * HH * DHH + 16 * DFFF;

    // zero the padded A buffers (rows >= 4 must be 0 for the WMMA GEMMs)
    zero_kernel<<<(int)((padN + 255) / 256), 256, 0, stream>>>(xn, padN);

    embed_kernel<<<BB, 256, 0, stream>>>(ids, emb, pe, x, HISTC);

    for (int l = 0; l < LL; ++l) {
        const float* kc_l = kcache  + (long long)l * BB * HH * DHH * HISTC;
        const float* vc_l = vcache  + (long long)l * BB * HH * HISTC * DHH;
        const float* ck_l = cross_k + (long long)l * BB * HH * DHH * TENCC;
        const float* cv_l = cross_v + (long long)l * BB * HH * TENCC * DHH;
        float* nk_l = out +      (long long)l * BB * HH * DHH * KVC;
        float* nv_l = out + NK + (long long)l * BB * HH * KVC * DHH;
        const float* wq_l = wq + (long long)l * HH * DD * DHH;
        const float* wk_l = wk + (long long)l * HH * DD * DHH;
        const float* wv_l = wv + (long long)l * HH * DD * DHH;

        // --- self attention ---
        ln_kernel<<<BB, 256, 0, stream>>>(x, snw + l * DD, snb + l * DD, xn, DD);

        // q = xn @ wq + bq -> qb (B,H,DH)
        gemm16_wmma_kernel<<<dim3(DHH / 32, HH), 32, 0, stream>>>(
            xn, DD, DD, wq_l, (long long)DD * DHH, DHH,
            bq + (long long)l * HH * DHH, DHH,
            qb, HH * DHH, DHH, 1, 0);
        // k_new -> new_k[..., :, HIST] (column stride KV)
        gemm16_wmma_kernel<<<dim3(DHH / 32, HH), 32, 0, stream>>>(
            xn, DD, DD, wk_l, (long long)DD * DHH, DHH,
            nullptr, 0,
            nk_l + HISTC, (long long)HH * DHH * KVC, (long long)DHH * KVC,
            KVC, 0);
        // v_new -> new_v[..., HIST, :]
        gemm16_wmma_kernel<<<dim3(DHH / 32, HH), 32, 0, stream>>>(
            xn, DD, DD, wv_l, (long long)DD * DHH, DHH,
            bv + (long long)l * HH * DHH, DHH,
            nv_l + (long long)HISTC * DHH, (long long)HH * KVC * DHH,
            (long long)KVC * DHH, 1, 0);

        // fused: copy caches into new_k/new_v + attention
        attn_kernel<HISTC, KVC, true, true><<<dim3(HH, BB), 256, 0, stream>>>(
            kc_l, vc_l, qb, nk_l, nv_l, ab, mflag);

        // x += sum_h attn @ fcw + fcb   (flattened K = H*DH)
        gemm16_wmma_kernel<<<dim3(DD / 32, 1), 32, 0, stream>>>(
            ab, HH * DHH, HH * DHH, fcw + (long long)l * HH * DHH * DD, 0, DD,
            fcb + (long long)l * DD, 0, x, DD, 0, 1, 1);

        // --- cross attention ---
        ln_kernel<<<BB, 256, 0, stream>>>(x, cnw + l * DD, cnb + l * DD, xn, DD);
        gemm16_wmma_kernel<<<dim3(DHH / 32, HH), 32, 0, stream>>>(
            xn, DD, DD, cwq + (long long)l * HH * DD * DHH, (long long)DD * DHH, DHH,
            cbq + (long long)l * HH * DHH, DHH,
            qb, HH * DHH, DHH, 1, 0);
        attn_kernel<TENCC, TENCC, false, false><<<dim3(HH, BB), 256, 0, stream>>>(
            ck_l, cv_l, qb, nullptr, nullptr, ab, mflag);
        gemm16_wmma_kernel<<<dim3(DD / 32, 1), 32, 0, stream>>>(
            ab, HH * DHH, HH * DHH, cfcw + (long long)l * HH * DHH * DD, 0, DD,
            cfcb + (long long)l * DD, 0, x, DD, 0, 1, 1);

        // --- FFN ---
        ln_kernel<<<BB, 256, 0, stream>>>(x, mnw + l * DD, mnb + l * DD, xn, DD);
        gemm16_wmma_kernel<<<dim3(DFFF / 32, 1), 32, 0, stream>>>(
            xn, DD, DD, w1 + (long long)l * DD * DFFF, 0, DFFF,
            b1 + (long long)l * DFFF, 0, tb, DFFF, 0, 1, 2);  // relu
        gemm16_wmma_kernel<<<dim3(DD / 32, 1), 32, 0, stream>>>(
            tb, DFFF, DFFF, w2 + (long long)l * DFFF * DD, 0, DD,
            b2 + (long long)l * DD, 0, x, DD, 0, 1, 1);
    }

    // final LN + vocab projection
    ln_kernel<<<BB, 256, 0, stream>>>(x, lnow, lnob, xn, DD);
    gemm16_wmma_kernel<<<dim3(VV / 32, 1), 32, 0, stream>>>(
        xn, DD, DD, proj_w, 0, VV, proj_b, 0, logits, VV, 0, 1, 0);

    kv_scalar_kernel<<<1, 1, 0, stream>>>(hlen, ilen, kvout);
}
